// PAM_Module_41145786695898
// MI455X (gfx1250) — matrix operand order repaired and checked by reading, not compiled, not run on hardware
//
#include <hip/hip_runtime.h>
#include <hip/hip_bf16.h>

// ---------------------------------------------------------------------------
// PAM module for MI455X (gfx1250, wave32, WMMA).
//
// Pipeline (all on `stream`, in-order):
//   pack x + weights -> bf16 (weights K-transposed [Co][K])
//   q branch: conv1x1 + 3x conv3x3(dil 1/3/6) -> qcat NHWC -> conv1x1 -> qT
//   k: conv3x3 twice (same weights) -> kbuf channel-major
//   v: conv1x1 -> vbuf channel-major
//   energy  = batched 64x64x64 GEMM (kbuf x qT)      -> ebuf fp32
//   softmax over channels                            -> aT bf16 (transposed)
//   out     = batched 64x64x64 GEMM (aT x vbuf)      -> obuf fp32 (aliases ebuf)
//   result  = gamma*out + x                          -> d_out fp32 NHWC
//
// All matmuls use v_wmma_f32_16x16x32_bf16 with fp32 accumulation.
// ---------------------------------------------------------------------------

#define NBATCH 8
#define HWSZ   4096            // 64*64
#define NPOS   32768           // NBATCH*HWSZ
#define CCH    256
#define CQCH   64

typedef __attribute__((ext_vector_type(8)))  unsigned short v8u16;
typedef __attribute__((ext_vector_type(16))) __bf16         v16bf;
typedef __attribute__((ext_vector_type(8)))  float          v8f;

union FragBF {
  struct { v8u16 lo, hi; } h;
  v16bf v;
};

__device__ __forceinline__ unsigned short f2bf(float f) {
  unsigned u = __float_as_uint(f);
  return (unsigned short)((u + 0x7FFFu + ((u >> 16) & 1u)) >> 16); // RNE
}

__device__ __forceinline__ v16bf frag_zero() {
  FragBF z;
  v8u16 t = {0, 0, 0, 0, 0, 0, 0, 0};
  z.h.lo = t; z.h.hi = t;
  return z.v;
}

// A fragment (16x32 bf16, M x K). rowp = &A[row_of_this_lane][k0], row-major.
// ISA layout: lanes 0-15 hold K 0..7 / 16..23, lanes 16-31 hold K 8..15 / 24..31.
__device__ __forceinline__ v16bf load_frag_a(const unsigned short* rowp, int lane) {
  const int ks = ((lane >> 4) & 1) << 3;     // 0 or 8 elements
  FragBF f;
  f.h.lo = *(const v8u16*)(rowp + ks);
  f.h.hi = *(const v8u16*)(rowp + ks + 16);
  return f.v;
}

// B fragment (32x16 bf16, K x N) from K-transposed storage Bt[n][k], row stride ldn.
// ISA layout: lane holds column n=lane%16; lanes 0-15 hold K 0..7 / 16..23, lanes 16-31 hold K 8..15 / 24..31.
__device__ __forceinline__ v16bf load_frag_bt(const unsigned short* base, int ldn, int lane) {
  const int n  = lane & 15;
  const int ks = ((lane >> 4) & 1) << 3;     // 0 or 8 elements
  const unsigned short* p = base + (size_t)n * ldn + ks;
  FragBF f;
  f.h.lo = *(const v8u16*)p;
  f.h.hi = *(const v8u16*)(p + 16);
  return f.v;
}

__device__ __forceinline__ v8f wmma_bf16(v16bf a, v16bf b, v8f c) {
  return __builtin_amdgcn_wmma_f32_16x16x32_bf16(false, a, false, b, (short)0, c,
                                                 false, false);
}

// Store one output element for conv kernels.
//  smode 0: NHWC bf16,  addr = row*ldc + c_off + col
//  smode 1: channel-major [b][Co][h][w]
//  smode 2: channel-major spatially transposed [b][Co][w][h]
__device__ __forceinline__ void store_conv(unsigned short* Out, float val,
                                           int rowg, int colg, int Co, int ldc,
                                           int c_off, int smode) {
  if (smode == 0) {
    Out[(size_t)rowg * ldc + c_off + colg] = f2bf(val);
  } else {
    const int b   = rowg >> 12;
    int pos       = rowg & 4095;
    if (smode == 2) pos = ((pos & 63) << 6) | (pos >> 6);
    Out[((size_t)(b * Co + colg) << 12) + pos] = f2bf(val);
  }
}

__device__ __forceinline__ void store_tile32(unsigned short* Out,
                                             const float* __restrict__ bias,
                                             v8f a00, v8f a01, v8f a10, v8f a11,
                                             int n0, int co0, int lane,
                                             int Co, int ldc, int c_off, int smode) {
  const int nlo = lane & 15;
  const int mhi = ((lane >> 4) & 1) << 3;    // C/D layout: lanes>=16 -> rows 8..15
  const float b0 = bias[co0 + nlo];
  const float b1 = bias[co0 + 16 + nlo];
#pragma unroll
  for (int r = 0; r < 8; ++r) {
    const int r0 = n0 + mhi + r;
    const int r1 = r0 + 16;
    store_conv(Out, a00[r] + b0, r0, co0 + nlo,      Co, ldc, c_off, smode);
    store_conv(Out, a01[r] + b1, r0, co0 + 16 + nlo, Co, ldc, c_off, smode);
    store_conv(Out, a10[r] + b0, r1, co0 + nlo,      Co, ldc, c_off, smode);
    store_conv(Out, a11[r] + b1, r1, co0 + 16 + nlo, Co, ldc, c_off, smode);
  }
}

// ----------------------------- pack kernels --------------------------------

__global__ void cvt_f32_bf16(unsigned short* __restrict__ dst,
                             const float* __restrict__ src, int n) {
  int i = blockIdx.x * blockDim.x + threadIdx.x;
  if (i < n) dst[i] = f2bf(src[i]);
}

// HWIO 1x1 [Cin][Co] -> [Co][Cin] bf16
__global__ void pack_w1x1(unsigned short* __restrict__ dst,
                          const float* __restrict__ src, int Cin, int Co) {
  int i = blockIdx.x * blockDim.x + threadIdx.x;
  if (i >= Cin * Co) return;
  int co = i / Cin, ci = i - co * Cin;
  dst[i] = f2bf(src[(size_t)ci * Co + co]);
}

// HWIO 3x3 [3][3][Cin][Co] -> [Co][9][Cin] bf16
__global__ void pack_w3x3(unsigned short* __restrict__ dst,
                          const float* __restrict__ src, int Cin, int Co) {
  int i = blockIdx.x * blockDim.x + threadIdx.x;
  if (i >= Co * 9 * Cin) return;
  int ci = i % Cin;
  int rest = i / Cin;
  int t = rest % 9;
  int co = rest / 9;
  dst[i] = f2bf(src[((size_t)t * Cin + ci) * Co + co]);
}

// ------------------------------ conv kernels -------------------------------

// Y[N,Co] = X[N,Cin] * W[Cin,Co] + bias.  One wave per 32x32 output tile.
__global__ void conv1x1_wmma(const unsigned short* __restrict__ X,
                             const unsigned short* __restrict__ Wt, // [Co][Cin]
                             const float* __restrict__ bias,
                             unsigned short* __restrict__ Out,
                             int Cin, int Co, int ldc, int c_off, int smode) {
  const int lane = threadIdx.x & 31;
  const int wave = blockIdx.x * (blockDim.x >> 5) + (threadIdx.x >> 5);
  const int colTiles = Co >> 5;
  const int rowTile  = wave / colTiles;
  const int colTile  = wave - rowTile * colTiles;
  const int n0  = rowTile << 5;
  const int co0 = colTile << 5;
  if (n0 >= NPOS) return;

  const int m = lane & 15;
  const unsigned short* a0p = X + (size_t)(n0 + m) * Cin;
  const unsigned short* a1p = X + (size_t)(n0 + 16 + m) * Cin;
  const unsigned short* b0p = Wt + (size_t)co0 * Cin;
  const unsigned short* b1p = Wt + (size_t)(co0 + 16) * Cin;

  v8f acc00 = {0.f,0.f,0.f,0.f,0.f,0.f,0.f,0.f};
  v8f acc01 = acc00, acc10 = acc00, acc11 = acc00;

#pragma unroll 4
  for (int k0 = 0; k0 < Cin; k0 += 32) {
    __builtin_prefetch(a0p + k0 + 128, 0, 1);   // global_prefetch_b8
    v16bf a0 = load_frag_a(a0p + k0, lane);
    v16bf a1 = load_frag_a(a1p + k0, lane);
    v16bf fb0 = load_frag_bt(b0p + k0, Cin, lane);
    v16bf fb1 = load_frag_bt(b1p + k0, Cin, lane);
    acc00 = wmma_bf16(a0, fb0, acc00);
    acc01 = wmma_bf16(a0, fb1, acc01);
    acc10 = wmma_bf16(a1, fb0, acc10);
    acc11 = wmma_bf16(a1, fb1, acc11);
  }
  store_tile32(Out, bias, acc00, acc01, acc10, acc11,
               n0, co0, lane, Co, ldc, c_off, smode);
}

// Implicit-GEMM 3x3 dilated conv, Cin=256, SAME padding.
__global__ void conv3x3_wmma(const unsigned short* __restrict__ X,  // NHWC bf16
                             const unsigned short* __restrict__ Wt, // [Co][9][256]
                             const float* __restrict__ bias,
                             unsigned short* __restrict__ Out,
                             int Co, int dil, int ldc, int c_off, int smode) {
  const int lane = threadIdx.x & 31;
  const int wave = blockIdx.x * (blockDim.x >> 5) + (threadIdx.x >> 5);
  const int colTiles = Co >> 5;
  const int rowTile  = wave / colTiles;
  const int colTile  = wave - rowTile * colTiles;
  const int n0  = rowTile << 5;   // 32 positions in one image row (32 | 64)
  const int co0 = colTile << 5;
  if (n0 >= NPOS) return;

  const int m  = lane & 15;
  const int b  = n0 >> 12;
  const int h  = (n0 >> 6) & 63;
  const int wm = (n0 & 63) + m;   // w coordinate of this lane's frag-0 row

  v8f acc00 = {0.f,0.f,0.f,0.f,0.f,0.f,0.f,0.f};
  v8f acc01 = acc00, acc10 = acc00, acc11 = acc00;

#pragma unroll
  for (int t = 0; t < 9; ++t) {
    const int hh = h + dil * (t / 3 - 1);
    if (hh < 0 || hh > 63) continue;          // whole tap OOB: uniform skip
    const int dw  = dil * (t % 3 - 1);
    const int wA0 = wm + dw;
    const int wA1 = wA0 + 16;
    const bool v0 = (unsigned)wA0 < 64u;
    const bool v1 = (unsigned)wA1 < 64u;
    const unsigned short* rowbase = X + ((size_t)((b * 64 + hh) * 64) * CCH);
    const unsigned short* a0p = rowbase + (size_t)wA0 * CCH;
    const unsigned short* a1p = rowbase + (size_t)wA1 * CCH;
    const unsigned short* b0p = Wt + ((size_t)co0 * 9 + t) * CCH;
    const unsigned short* b1p = Wt + ((size_t)(co0 + 16) * 9 + t) * CCH;
#pragma unroll 2
    for (int k0 = 0; k0 < CCH; k0 += 32) {
      v16bf a0 = v0 ? load_frag_a(a0p + k0, lane) : frag_zero();
      v16bf a1 = v1 ? load_frag_a(a1p + k0, lane) : frag_zero();
      v16bf fb0 = load_frag_bt(b0p + k0, 9 * CCH, lane);
      v16bf fb1 = load_frag_bt(b1p + k0, 9 * CCH, lane);
      acc00 = wmma_bf16(a0, fb0, acc00);
      acc01 = wmma_bf16(a0, fb1, acc01);
      acc10 = wmma_bf16(a1, fb0, acc10);
      acc11 = wmma_bf16(a1, fb1, acc11);
    }
  }
  store_tile32(Out, bias, acc00, acc01, acc10, acc11,
               n0, co0, lane, Co, ldc, c_off, smode);
}

// --------------------- batched 64x64x64 GEMM (einsums) ---------------------
// 2048 independent matrices; one wave each. C = A(64x64) * Bt^T(64x64), fp32 out.
__global__ void bgemm64_wmma(const unsigned short* __restrict__ A,
                             const unsigned short* __restrict__ Bt,
                             float* __restrict__ C) {
  const int lane = threadIdx.x & 31;
  const int mat  = blockIdx.x * (blockDim.x >> 5) + (threadIdx.x >> 5);
  if (mat >= NBATCH * CCH) return;
  const unsigned short* Ab = A  + ((size_t)mat << 12);
  const unsigned short* Bb = Bt + ((size_t)mat << 12);
  float* Cb = C + ((size_t)mat << 12);
  const int m = lane & 15;

  v8f z = {0.f,0.f,0.f,0.f,0.f,0.f,0.f,0.f};
  v8f acc[4][4];
#pragma unroll
  for (int i = 0; i < 4; ++i)
#pragma unroll
    for (int j = 0; j < 4; ++j) acc[i][j] = z;

#pragma unroll
  for (int k0 = 0; k0 < 64; k0 += 32) {
    v16bf af[4], bf[4];
#pragma unroll
    for (int f = 0; f < 4; ++f) {
      af[f] = load_frag_a(Ab + (size_t)((f << 4) + m) * 64 + k0, lane);
      bf[f] = load_frag_bt(Bb + ((size_t)(f << 4)) * 64 + k0, 64, lane);
    }
#pragma unroll
    for (int i = 0; i < 4; ++i)
#pragma unroll
      for (int j = 0; j < 4; ++j) acc[i][j] = wmma_bf16(af[i], bf[j], acc[i][j]);
  }

  const int nlo = lane & 15;
  const int mhi = ((lane >> 4) & 1) << 3;
#pragma unroll
  for (int i = 0; i < 4; ++i)
#pragma unroll
    for (int j = 0; j < 4; ++j)
#pragma unroll
      for (int r = 0; r < 8; ++r)
        Cb[(size_t)((i << 4) + mhi + r) * 64 + (j << 4) + nlo] = acc[i][j][r];
}

// -------------------------- softmax over channels --------------------------
// attention[b,i,k,c] = softmax_c(energy[b,i,k,c]); stored transposed:
// aT[b][c][k][i] so GEMM2's A operand is row-major.
__global__ void softmax_c(const float* __restrict__ E,       // [b][c][4096]
                          unsigned short* __restrict__ AT) { // [b][c][4096]
  const int idx = blockIdx.x * blockDim.x + threadIdx.x;     // over B*4096
  if (idx >= NBATCH * HWSZ) return;
  const int b   = idx >> 12;
  const int pos = idx & 4095;
  const float* e = E + ((size_t)b * CCH << 12) + pos;

  float mx = -3.4e38f;
  for (int c = 0; c < CCH; ++c) mx = fmaxf(mx, e[(size_t)c << 12]);
  float s = 0.f;
  for (int c = 0; c < CCH; ++c) s += __expf(e[(size_t)c << 12] - mx);
  const float inv = 1.f / s;

  const int tpos = ((pos & 63) << 6) | (pos >> 6);           // [i][k] -> [k][i]
  unsigned short* o = AT + ((size_t)b * CCH << 12) + tpos;
  for (int c = 0; c < CCH; ++c)
    o[(size_t)c << 12] = f2bf(__expf(e[(size_t)c << 12] - mx) * inv);
}

// ------------------------------- epilogue ----------------------------------
__global__ void final_add(const float* __restrict__ O,   // [b][c][4096]
                          const float* __restrict__ x,   // NHWC fp32
                          const float* __restrict__ gamma,
                          float* __restrict__ out) {
  const int idx = blockIdx.x * blockDim.x + threadIdx.x;
  if (idx >= NPOS * CCH) return;
  const int c   = idx & 255;
  const int pos = (idx >> 8) & 4095;
  const int b   = idx >> 20;
  const float o = O[((size_t)(b * CCH + c) << 12) + pos];
  out[idx] = gamma[0] * o + x[idx];
}

// ---------------------------------------------------------------------------
extern "C" void kernel_launch(void* const* d_in, const int* in_sizes, int n_in,
                              void* d_out, int out_size, void* d_ws, size_t ws_size,
                              hipStream_t stream) {
  (void)in_sizes; (void)n_in; (void)out_size; (void)ws_size;
  const float* x    = (const float*)d_in[0];
  const float* wq   = (const float*)d_in[1];
  const float* bq   = (const float*)d_in[2];
  const float* wq1  = (const float*)d_in[3];
  const float* bq1  = (const float*)d_in[4];
  const float* wq2  = (const float*)d_in[5];
  const float* bq2  = (const float*)d_in[6];
  const float* wq3  = (const float*)d_in[7];
  const float* bq3  = (const float*)d_in[8];
  const float* wq4  = (const float*)d_in[9];
  const float* bq4  = (const float*)d_in[10];
  const float* wk   = (const float*)d_in[11];
  const float* bk   = (const float*)d_in[12];
  const float* wv   = (const float*)d_in[13];
  const float* bv   = (const float*)d_in[14];
  const float* gam  = (const float*)d_in[15];
  float* out = (float*)d_out;
  char* ws = (char*)d_ws;

  // ---- workspace arena (all offsets 256B-aligned); total ~137 MB ----
  size_t off = 0;
  unsigned short* xb   = (unsigned short*)(ws + off); off += (size_t)NPOS * CCH * 2;
  unsigned short* wqT  = (unsigned short*)(ws + off); off += (size_t)CQCH * CCH * 2;
  unsigned short* wq1T = (unsigned short*)(ws + off); off += (size_t)CQCH * 9 * CCH * 2;
  unsigned short* wq2T = (unsigned short*)(ws + off); off += (size_t)CQCH * 9 * CCH * 2;
  unsigned short* wq3T = (unsigned short*)(ws + off); off += (size_t)CQCH * 9 * CCH * 2;
  unsigned short* wq4T = (unsigned short*)(ws + off); off += (size_t)CCH * CCH * 2;
  unsigned short* wkT  = (unsigned short*)(ws + off); off += (size_t)CCH * 9 * CCH * 2;
  unsigned short* wvT  = (unsigned short*)(ws + off); off += (size_t)CCH * CCH * 2;
  unsigned short* qcat = (unsigned short*)(ws + off); off += (size_t)NPOS * CCH * 2;
  unsigned short* qT   = (unsigned short*)(ws + off); off += (size_t)NPOS * CCH * 2;
  unsigned short* kbuf = (unsigned short*)(ws + off); off += (size_t)NPOS * CCH * 2;
  unsigned short* vbuf = (unsigned short*)(ws + off); off += (size_t)NPOS * CCH * 2;
  unsigned short* aT   = (unsigned short*)(ws + off); off += (size_t)NPOS * CCH * 2;
  float* ebuf          = (float*)(ws + off);           // 33.5 MB fp32
  unsigned short* k1 = qcat;   // qcat is dead after the q4 conv -> reuse for k1
  float* obuf = ebuf;          // ebuf is dead after softmax -> reuse for out

  // ---- pack to bf16 ----
  cvt_f32_bf16<<<(NPOS * CCH + 255) / 256, 256, 0, stream>>>(xb, x, NPOS * CCH);
  pack_w1x1<<<(CCH * CQCH + 255) / 256, 256, 0, stream>>>(wqT,  wq,  CCH, CQCH);
  pack_w3x3<<<(CQCH * 9 * CCH + 255) / 256, 256, 0, stream>>>(wq1T, wq1, CCH, CQCH);
  pack_w3x3<<<(CQCH * 9 * CCH + 255) / 256, 256, 0, stream>>>(wq2T, wq2, CCH, CQCH);
  pack_w3x3<<<(CQCH * 9 * CCH + 255) / 256, 256, 0, stream>>>(wq3T, wq3, CCH, CQCH);
  pack_w1x1<<<(CCH * CCH + 255) / 256, 256, 0, stream>>>(wq4T, wq4, CCH, CCH);
  pack_w3x3<<<(CCH * 9 * CCH + 255) / 256, 256, 0, stream>>>(wkT, wk, CCH, CCH);
  pack_w1x1<<<(CCH * CCH + 255) / 256, 256, 0, stream>>>(wvT, wv, CCH, CCH);

  // ---- q branch -> qcat (NHWC, channel offsets 0/64/128/192) ----
  conv1x1_wmma<<<512, 128, 0, stream>>>(xb, wqT,  bq,  qcat, CCH, CQCH, CCH, 0,   0);
  conv3x3_wmma<<<512, 128, 0, stream>>>(xb, wq1T, bq1, qcat, CQCH, 1, CCH, 64,  0);
  conv3x3_wmma<<<512, 128, 0, stream>>>(xb, wq2T, bq2, qcat, CQCH, 3, CCH, 128, 0);
  conv3x3_wmma<<<512, 128, 0, stream>>>(xb, wq3T, bq3, qcat, CQCH, 6, CCH, 192, 0);
  // q4: qT[b][c][k][i] = q[b,i,k,c]  (smode 2: channel-major + spatial transpose)
  conv1x1_wmma<<<2048, 128, 0, stream>>>(qcat, wq4T, bq4, qT, CCH, CCH, CCH, 0, 2);

  // ---- k (double conv, same weights) and v ----
  conv3x3_wmma<<<2048, 128, 0, stream>>>(xb, wkT, bk, k1,   CCH, 1, CCH, 0, 0);
  conv3x3_wmma<<<2048, 128, 0, stream>>>(k1, wkT, bk, kbuf, CCH, 1, CCH, 0, 1);
  conv1x1_wmma<<<2048, 128, 0, stream>>>(xb, wvT, bv, vbuf, CCH, CCH, CCH, 0, 1);

  // ---- energy = K_c @ Q_c per (b,c); softmax over c; out = Att_c^T @ V_c^T ----
  bgemm64_wmma<<<512, 128, 0, stream>>>(kbuf, qT, ebuf);
  softmax_c<<<(NBATCH * HWSZ + 255) / 256, 256, 0, stream>>>(ebuf, aT);
  bgemm64_wmma<<<512, 128, 0, stream>>>(aT, vbuf, obuf);

  // ---- gamma*out + x ----
  final_add<<<(NPOS * CCH + 255) / 256, 256, 0, stream>>>(obuf, x, gam, out);
}
